// Decoder_41644002902672
// MI455X (gfx1250) — matrix operand order repaired and checked by reading, not compiled, not run on hardware
//
#include <hip/hip_runtime.h>
#include <hip/hip_bf16.h>
#include <stdint.h>

// ---------------- problem constants (match reference) ----------------
#define G_        2
#define Q_        8
#define C_        1024
#define D_        8
#define DIMS_     1024
#define B_        4
#define L_        4096
#define LOUT_     8192              // 2*L
#define HEADO_    512
#define ZPAD_ROWS (L_ + 2)          // z with 1 zero halo row each side
#define ZUPAD_ROWS (LOUT_ + 6)      // zu with 3 zero halo rows each side

typedef __attribute__((ext_vector_type(16))) __bf16         v16bf;
typedef __attribute__((ext_vector_type(8)))  float          v8f;
typedef __attribute__((ext_vector_type(8)))  unsigned short ush8;
typedef __attribute__((ext_vector_type(16))) unsigned short ush16;

// CDNA5 async global->LDS path (ASYNCcnt-tracked), with manual fallback.
// Host pass: __has_builtin is false -> fallback, so no AS casts on x86.
// Builtin params are b128 payload pointers: (vector_size(16) int)* in AS1/AS3.
#if __has_builtin(__builtin_amdgcn_global_load_async_to_lds_b128) && \
    __has_builtin(__builtin_amdgcn_s_wait_asynccnt)
#define USE_ASYNC_LDS 1
typedef int v4i_ __attribute__((vector_size(16)));
typedef __attribute__((address_space(1))) v4i_ GlobV4;   // global
typedef __attribute__((address_space(3))) v4i_ LdsV4;    // LDS
#else
#define USE_ASYNC_LDS 0
#endif

static __device__ __forceinline__ unsigned short f2bf(float f) {
  unsigned u = __float_as_uint(f);
  u += 0x7fffu + ((u >> 16) & 1u);          // round-to-nearest-even
  return (unsigned short)(u >> 16);
}

// ---------------- stage 1+2: dequant + per-group projection ----------------
__global__ __launch_bounds__(256) void dequant_project(
    const int* __restrict__ idx,        // [B, G*Q, L]
    const float* __restrict__ cb,       // [G,Q,C,D]
    const float* __restrict__ sc,       // [G,Q,D]
    const float* __restrict__ Wo,       // [G,D,512]
    const float* __restrict__ bo,       // [G,512]
    unsigned short* __restrict__ zpad)  // [B, L+2, 1024] bf16
{
  const int t = blockIdx.x * blockDim.x + threadIdx.x;   // 16384 threads
  const int b = t >> 12;
  const int l = t & (L_ - 1);
  float summed[G_][D_];
  for (int g = 0; g < G_; ++g)
    for (int d = 0; d < D_; ++d) summed[g][d] = 0.f;

  for (int g = 0; g < G_; ++g) {
    for (int q = 0; q < Q_; ++q) {
      const int gq = g * Q_ + q;
      const int code = idx[((size_t)b * (G_ * Q_) + gq) * L_ + l];
      if (code < 0) continue;                       // dropout-padded (-1)
      const float* cbr = cb + ((size_t)gq * C_ + code) * D_;
      const float* scr = sc + (size_t)gq * D_;
      for (int d = 0; d < D_; ++d) summed[g][d] += scr[d] * cbr[d];
    }
  }

  unsigned short* zrow = zpad + ((size_t)b * ZPAD_ROWS + 1 + l) * DIMS_;
  for (int g = 0; g < G_; ++g) {
    const float* W  = Wo + (size_t)g * D_ * 512;    // wave-uniform -> scalar loads
    const float* bb = bo + (size_t)g * 512;
    for (int e = 0; e < 512; ++e) {
      float a = bb[e];
      for (int d = 0; d < D_; ++d) a += summed[g][d] * W[d * 512 + e];
      zrow[g * 512 + e] = f2bf(a);
    }
  }
}

// ---------------- weight packing into WMMA-B fragment-native layout ----------
// Packed: [ktile][ntiles_total][lane(32)][16] bf16.  32x16 B tile:
// lane<16 -> N=lane, K=0..15 ; lane>=16 -> N=lane-16, K=16..31.
__global__ __launch_bounds__(256) void pack_wup(
    const float* __restrict__ Wup,      // IOH [1024,1024,4]
    unsigned short* __restrict__ out)   // [2][64][64][32][16]
{
  const size_t t = (size_t)blockIdx.x * blockDim.x + threadIdx.x;  // 2*2048*1024
  const int n = (int)(t & 1023);
  const int k = (int)((t >> 10) & 2047);
  const int p = (int)(t >> 21);                      // parity 0=even,1=odd
  const int i = k & 1023;
  const int tap = (k < 1024) ? p : (p + 2);          // even: taps 0,2; odd: 1,3
  const float w = Wup[((size_t)i * DIMS_ + n) * 4 + tap];
  const int ktile = k >> 5, kin = k & 31, ntile = n >> 4, nin = n & 15;
  const int lane = (kin < 16) ? nin : (16 + nin);
  const int elem = kin & 15;
  const size_t o = ((((size_t)p * 64 + ktile) * 64 + ntile) * 32 + lane) * 16 + elem;
  out[o] = f2bf(w);
}

__global__ __launch_bounds__(256) void pack_whead(
    const float* __restrict__ Wh,       // OIH [512,1024,7]
    unsigned short* __restrict__ out)   // [224][32][32][16]
{
  const size_t t = (size_t)blockIdx.x * blockDim.x + threadIdx.x;  // 7168*512
  const int n  = (int)(t & 511);
  const int kk = (int)(t >> 9);                      // 0..7167 = tap*1024 + c
  const int tap = kk >> 10, c = kk & 1023;
  const float w = Wh[((size_t)n * DIMS_ + c) * 7 + tap];
  const int ktile = kk >> 5, kin = kk & 31, ntile = n >> 4, nin = n & 15;
  const int lane = (kin < 16) ? nin : (16 + nin);
  const int elem = kin & 15;
  const size_t o = (((size_t)ktile * 32 + ntile) * 32 + lane) * 16 + elem;
  out[o] = f2bf(w);
}

// ---------------- zero the halo rows (ws is poisoned by the harness) --------
__global__ __launch_bounds__(256) void zero_halos(
    unsigned short* __restrict__ zpad, unsigned short* __restrict__ zupad)
{
  const int t = blockIdx.x * blockDim.x + threadIdx.x;   // 32768
  if (t < 8192) {
    const int b = t >> 11, r = (t >> 10) & 1, c = t & 1023;
    zpad[((size_t)b * ZPAD_ROWS + (size_t)r * (ZPAD_ROWS - 1)) * DIMS_ + c] = 0;
  } else {
    const int u = t - 8192;                              // 24576
    const int b = u / 6144, rem = u % 6144;
    const int r6 = rem >> 10, c = rem & 1023;
    const int row = (r6 < 3) ? r6 : (ZUPAD_ROWS - 6 + r6);
    zupad[((size_t)b * ZUPAD_ROWS + row) * DIMS_ + c] = 0;
  }
}

// ---------------- WMMA GEMM core -------------------------------------------
// C[M,N] += A[M,K]*B[K,N]; A rows are overlapping contiguous windows
// (row m at Abase + m*1024 elements, length K).  256 threads = 8 wave32,
// block tile 128x128, wave tile 32x64 (2x4 v8f accumulators).
// A double-buffered through LDS (80B row pitch); B fragment-packed in L2.
struct AccTile { v8f a[2][4]; };

template <int KTOT, int NTILES_TOTAL>
static __device__ __forceinline__ void gemm_core(
    const unsigned short* __restrict__ Abase,
    int m0,
    const unsigned short* __restrict__ Bpack,
    int ntile0,
    AccTile& acc)
{
  __shared__ unsigned short As[2][128][40];            // ping-pong, 2x10KB
  const int tid  = threadIdx.x;
  const int lane = tid & 31;
  const int wid  = tid >> 5;
  const int wm   = wid & 3;                            // 4 waves along M
  const int wn   = wid >> 2;                           // 2 waves along N

  v8f vzero;
  for (int i = 0; i < 8; ++i) vzero[i] = 0.f;
  for (int mi = 0; mi < 2; ++mi)
    for (int ni = 0; ni < 4; ++ni) acc.a[mi][ni] = vzero;

  const int arow  = tid >> 1;                          // 128 rows, 2 thr/row
  const int ahalf = tid & 1;                           // 16 bf16 each
  const unsigned short* asrc =
      Abase + (size_t)(m0 + arow) * 1024 + ahalf * 16;

  constexpr int KT = KTOT / 32;

#if USE_ASYNC_LDS
  // ---- async global->LDS (ASYNCcnt) pipeline ----
  auto issue_tile = [&](int kt, int buf) {
    const unsigned short* g0 = asrc + kt * 32;
    unsigned short* l0 = &As[buf][arow][ahalf * 16];
    __builtin_amdgcn_global_load_async_to_lds_b128(
        (GlobV4*)g0,       (LdsV4*)l0,       0, 0);
    __builtin_amdgcn_global_load_async_to_lds_b128(
        (GlobV4*)(g0 + 8), (LdsV4*)(l0 + 8), 0, 0);
  };
  issue_tile(0, 0);
  __builtin_amdgcn_s_wait_asynccnt(0);
  __syncthreads();
#else
  // ---- manual prologue: tile 0 -> buf 0 ----
  {
    const uint4* s = (const uint4*)asrc;
    const uint4 v0 = s[0];
    const uint4 v1 = s[1];
    *(uint4*)&As[0][arow][ahalf * 16 + 0] = v0;
    *(uint4*)&As[0][arow][ahalf * 16 + 8] = v1;
  }
  __syncthreads();
#endif

  for (int kt = 0; kt < KT; ++kt) {
    const int buf = kt & 1;

#if USE_ASYNC_LDS
    if (kt + 1 < KT) issue_tile(kt + 1, buf ^ 1);      // overlaps with compute
#else
    uint4 n0{}, n1{};
    if (kt + 1 < KT) {
      const uint4* s = (const uint4*)(asrc + (kt + 1) * 32);
      n0 = s[0];
      n1 = s[1];
      if (kt + 2 < KT) __builtin_prefetch(asrc + (kt + 2) * 32, 0, 1);
    }
#endif

    // B fragments first (one load clause; drains under the WMMAs)
    v16bf bfrag[4];
#pragma unroll
    for (int ni = 0; ni < 4; ++ni) {
      const int ntile = ntile0 + wn * 4 + ni;
      const ush16* bp = (const ush16*)(
          Bpack + (((size_t)kt * NTILES_TOTAL + ntile) * 32 + lane) * 16);
      bfrag[ni] = __builtin_bit_cast(v16bf, *bp);
    }

    // A fragments (contiguous K layout matching B pack: lane<16 K0-15, lane>=16 K16-31)
    v16bf afrag[2];
    const int hoff = (lane >> 4) * 16;
#pragma unroll
    for (int mi = 0; mi < 2; ++mi) {
      const int row = wm * 32 + mi * 16 + (lane & 15);
      const ush8 lo = *(const ush8*)&As[buf][row][hoff];
      const ush8 hi = *(const ush8*)&As[buf][row][hoff + 8];
      const ush16 cc = __builtin_shufflevector(
          lo, hi, 0, 1, 2, 3, 4, 5, 6, 7, 8, 9, 10, 11, 12, 13, 14, 15);
      afrag[mi] = __builtin_bit_cast(v16bf, cc);
    }

#pragma unroll
    for (int ni = 0; ni < 4; ++ni)
#pragma unroll
      for (int mi = 0; mi < 2; ++mi)
        acc.a[mi][ni] = __builtin_amdgcn_wmma_f32_16x16x32_bf16(
            false, afrag[mi], false, bfrag[ni], (short)0, acc.a[mi][ni],
            false, false);

#if USE_ASYNC_LDS
    if (kt + 1 < KT) {
      __builtin_amdgcn_s_wait_asynccnt(0);             // tile kt+1 landed
      __syncthreads();                                 // publish to all waves
    }
#else
    if (kt + 1 < KT) {
      *(uint4*)&As[buf ^ 1][arow][ahalf * 16 + 0] = n0;
      *(uint4*)&As[buf ^ 1][arow][ahalf * 16 + 8] = n1;
      __syncthreads();                                 // single barrier/iter
    }
#endif
  }
}

// ---------------- stage 3: ConvTranspose1d as parity-split GEMM -------------
__global__ __launch_bounds__(256) void gemm_convtrans(
    const unsigned short* __restrict__ zpad,   // [B, L+2, 1024] bf16
    const unsigned short* __restrict__ wpack,  // [2][64][64][32][16] bf16
    const float* __restrict__ bup,             // [1024]
    unsigned short* __restrict__ zupad)        // [B, 2L+6, 1024] bf16
{
  const int bz = blockIdx.z;                   // B*2
  const int b = bz >> 1, p = bz & 1;           // parity
  const int m0 = blockIdx.y * 128;             // s tile
  const int ntile0 = blockIdx.x * 8;
  // zu[2s+p] = z[s-1+p]*W_p + z[s+p]*W_{p+2}; window = padded rows s+p, s+p+1
  const unsigned short* Abase =
      zpad + ((size_t)b * ZPAD_ROWS + p) * DIMS_;
  const unsigned short* Bp = wpack + (size_t)p * (64ull * 64 * 32 * 16);

  AccTile acc;
  gemm_core<2048, 64>(Abase, m0, Bp, ntile0, acc);

  const int lane = threadIdx.x & 31, wid = threadIdx.x >> 5;
  const int wm = wid & 3, wn = wid >> 2;
  unsigned short* zb = zupad + (size_t)b * ZUPAD_ROWS * DIMS_;
#pragma unroll
  for (int ni = 0; ni < 4; ++ni) {
    const int n = (ntile0 + wn * 4 + ni) * 16 + (lane & 15);
    const float bias = bup[n];
#pragma unroll
    for (int mi = 0; mi < 2; ++mi) {
      const int sbase = m0 + wm * 32 + mi * 16 + 8 * (lane >> 4);
#pragma unroll
      for (int r = 0; r < 8; ++r) {
        const int orow = 3 + 2 * (sbase + r) + p;       // halo offset 3
        zb[(size_t)orow * DIMS_ + n] = f2bf(acc.a[mi][ni][r] + bias);
      }
    }
  }
}

// ---------------- stage 4: head Conv1d(k=7) as windowed GEMM ----------------
__global__ __launch_bounds__(256) void gemm_head(
    const unsigned short* __restrict__ zupad,  // [B, 2L+6, 1024] bf16
    const unsigned short* __restrict__ wpack,  // [224][32][32][16] bf16
    const float* __restrict__ bh,              // [512]
    float* __restrict__ out)                   // [B, 512, 8192] fp32
{
  const int b = blockIdx.z;
  const int m0 = blockIdx.y * 128;             // t tile
  const int ntile0 = blockIdx.x * 8;
  // row t -> padded rows t..t+6 == zu[t-3 .. t+3]
  const unsigned short* Abase = zupad + (size_t)b * ZUPAD_ROWS * DIMS_;

  AccTile acc;
  gemm_core<7168, 32>(Abase, m0, wpack, ntile0, acc);

  const int lane = threadIdx.x & 31, wid = threadIdx.x >> 5;
  const int wm = wid & 3, wn = wid >> 2;
#pragma unroll
  for (int ni = 0; ni < 4; ++ni) {
    const int o = (ntile0 + wn * 4 + ni) * 16 + (lane & 15);
    const float bias = bh[o];
    float* ob = out + ((size_t)b * HEADO_ + o) * LOUT_;
    const int t0 = m0 + wm * 32 + 8 * (lane >> 4);     // 8 contiguous t per lane
#pragma unroll
    for (int mi = 0; mi < 2; ++mi) {
      const v8f c = acc.a[mi][ni];
      const int tt = t0 + mi * 16;
      *(float4*)(ob + tt)     = make_float4(c[0] + bias, c[1] + bias,
                                            c[2] + bias, c[3] + bias);
      *(float4*)(ob + tt + 4) = make_float4(c[4] + bias, c[5] + bias,
                                            c[6] + bias, c[7] + bias);
    }
  }
}

// ---------------- host entry -------------------------------------------------
extern "C" void kernel_launch(void* const* d_in, const int* in_sizes, int n_in,
                              void* d_out, int out_size, void* d_ws, size_t ws_size,
                              hipStream_t stream) {
  (void)in_sizes; (void)n_in; (void)out_size; (void)ws_size;
  const int*   indices   = (const int*)  d_in[0];   // harness: integer -> int*
  const float* codebooks = (const float*)d_in[1];
  const float* scales    = (const float*)d_in[2];
  const float* W_out     = (const float*)d_in[3];
  const float* b_out     = (const float*)d_in[4];
  const float* W_up      = (const float*)d_in[5];
  const float* b_up      = (const float*)d_in[6];
  const float* W_head    = (const float*)d_in[7];
  const float* b_head    = (const float*)d_in[8];

  char* ws = (char*)d_ws;
  size_t off = 0;
  unsigned short* zpad  = (unsigned short*)(ws + off);
  off += (size_t)B_ * ZPAD_ROWS * DIMS_ * 2;                    // 33.6 MB
  unsigned short* zupad = (unsigned short*)(ws + off);
  off += (size_t)B_ * ZUPAD_ROWS * DIMS_ * 2;                   // 67.2 MB
  unsigned short* wup_p = (unsigned short*)(ws + off);
  off += (size_t)2 * 2048 * 1024 * 2;                           // 8.4 MB
  unsigned short* wh_p  = (unsigned short*)(ws + off);
  off += (size_t)7168 * 512 * 2;                                // 7.3 MB

  pack_wup  <<<16384, 256, 0, stream>>>(W_up, wup_p);
  pack_whead<<<14336, 256, 0, stream>>>(W_head, wh_p);
  zero_halos<<<128,   256, 0, stream>>>(zpad, zupad);
  dequant_project<<<64, 256, 0, stream>>>(indices, codebooks, scales,
                                          W_out, b_out, zpad);
  gemm_convtrans<<<dim3(8, 32, 8), 256, 0, stream>>>(zpad, wup_p, b_up, zupad);
  gemm_head     <<<dim3(4, 64, 4), 256, 0, stream>>>(zupad, wh_p, b_head,
                                                     (float*)d_out);
}
